// VAEDecoder_31705448579500
// MI455X (gfx1250) — matrix operand order repaired and checked by reading, not compiled, not run on hardware
//
#include <hip/hip_runtime.h>
#include <hip/hip_bf16.h>

#define B_   2048
#define T_   120
#define V_   64
#define E_   128
#define H_   512
#define LAT_ 256
#define G3_  1536   // 3*H

typedef __attribute__((ext_vector_type(16))) __bf16 v16bf;
typedef __attribute__((ext_vector_type(8)))  __bf16 v8bf;
typedef __attribute__((ext_vector_type(8)))  float  v8f;

#define SRH 520    // bf16 row stride for h buffers (pad vs 512 to dodge bank conflicts)
#define SRG 1544   // f32 row stride for gate buffer (pad vs 1536)

// ---------------- setup kernels ----------------

// fp32 -> bf16 weight conversion (row-major (N,K) kept; B-fragments read rows)
__global__ void k_convert(const float* Whh0, const float* Wih1, const float* Whh1,
                          const float* Wout,
                          __bf16* wHH0, __bf16* wIH1, __bf16* wHH1, __bf16* wOUT) {
  int i = blockIdx.x * blockDim.x + threadIdx.x;
  if (i < G3_ * H_) {
    wHH0[i] = (__bf16)Whh0[i];
    wIH1[i] = (__bf16)Wih1[i];
    wHH1[i] = (__bf16)Whh1[i];
  }
  if (i < V_ * H_) wOUT[i] = (__bf16)Wout[i];
}

// Gtab[v][n] = b_ih0[n] + sum_e embed[v][e] * W_ih0[n][e]   (64 x 1536)
__global__ void k_gtab(const float* embed, const float* Wih0, const float* bih0,
                       float* Gtab) {
  int i = blockIdx.x * blockDim.x + threadIdx.x;
  if (i >= V_ * G3_) return;
  int v = i / G3_, n = i - v * G3_;
  float acc = bih0[n];
  const float* er = embed + v * E_;
  const float* wr = Wih0 + (size_t)n * (E_ + LAT_);
  #pragma unroll 4
  for (int e = 0; e < E_; ++e) acc += er[e] * wr[e];
  Gtab[i] = acc;
}

// gz[b][n] = sum_k z[b][k] * W_ih0[n][128+k]   (2048 x 1536)
__global__ void k_gz(const float* z, const float* Wih0, float* gz) {
  int i = blockIdx.x * blockDim.x + threadIdx.x;
  if (i >= B_ * G3_) return;
  int b = i / G3_, n = i - b * G3_;
  float acc = 0.f;
  const float* zr = z + b * LAT_;
  const float* wr = Wih0 + (size_t)n * (E_ + LAT_) + E_;
  #pragma unroll 4
  for (int k = 0; k < LAT_; ++k) acc += zr[k] * wr[k];
  gz[i] = acc;
}

// initial hidden states: hid = z @ fc_hidden_w^T + b ;  h0 = hid[:, :512], h1 = hid[:, 512:]
__global__ void k_hinit(const float* z, const float* fhw, const float* fhb,
                        float* h0i, float* h1i) {
  int i = blockIdx.x * blockDim.x + threadIdx.x;
  if (i >= B_ * 2 * H_) return;
  int b = i >> 10, j = i & 1023;
  float acc = fhb[j];
  const float* zr = z + b * LAT_;
  const float* wr = fhw + (size_t)j * LAT_;
  #pragma unroll 4
  for (int k = 0; k < LAT_; ++k) acc += zr[k] * wr[k];
  if (j < H_) h0i[b * H_ + j]         = acc;
  else        h1i[b * H_ + (j - H_)]  = acc;
}

// ---------------- main persistent GRU kernel ----------------

// Accumulate  acc[12] += A(16x512 bf16 in LDS, row stride SRH) @ W[nb..nb+191, :]^T
// A-fragment layout used here (contiguous K; B uses the same order):
//   lane<16 : row=lane,    K = kk..kk+15
//   lane>=16: row=lane-16, K = kk+16..kk+31
// B-fragment: lane column n = n0+(lane&15); 16 contiguous K from W row, offset half*16.
__device__ __forceinline__ void gemm12(const __bf16* __restrict__ W,
                                       const __bf16* hb, int nb, int lane,
                                       v8f acc[12]) {
  const int half = lane >> 4;
  const int row  = lane & 15;
  #pragma unroll 1
  for (int kk = 0; kk < H_; kk += 32) {
    const int k1 = kk + half * 16;
    v8bf lo = *(const v8bf*)(hb + row * SRH + k1);
    v8bf hi = *(const v8bf*)(hb + row * SRH + k1 + 8);
    v16bf a;
    #pragma unroll
    for (int i = 0; i < 8; ++i) { a[i] = lo[i]; a[8 + i] = hi[i]; }
    const int kb = kk + half * 16;
    #pragma unroll
    for (int j = 0; j < 12; ++j) {
      v16bf bf = *(const v16bf*)(W + (size_t)(nb + j * 16 + row) * H_ + kb);
      acc[j] = __builtin_amdgcn_wmma_f32_16x16x32_bf16(
          false, a, false, bf, (short)0, acc[j], false, false);
    }
  }
}

__global__ __launch_bounds__(256) void k_gru(
    const int* __restrict__ target,
    const float* __restrict__ Gtab, const float* __restrict__ gz,
    const __bf16* __restrict__ wHH0, const __bf16* __restrict__ wIH1,
    const __bf16* __restrict__ wHH1, const __bf16* __restrict__ wOUT,
    const float* __restrict__ bhh0, const float* __restrict__ bih1,
    const float* __restrict__ bhh1, const float* __restrict__ bout,
    const float* __restrict__ h0i, const float* __restrict__ h1i,
    float* __restrict__ out) {
  __shared__ __align__(32) float  gates[16 * SRG];   // 98.8 KB
  __shared__ __align__(32) __bf16 hb0[16 * SRH];     // 16.6 KB  (h0 state, bf16)
  __shared__ __align__(32) __bf16 hb1[16 * SRH];     // 16.6 KB  (h1 state, bf16)
  __shared__ __align__(32) __bf16 gi1n[16 * SRH];    // 16.6 KB  (ih n-gate spill)
  __shared__ int stok[16];

  const int tid  = threadIdx.x;
  const int lane = tid & 31;
  const int wave = tid >> 5;
  const int b0   = blockIdx.x * 16;     // 16 batch rows per WG, 128 WGs
  const int half = lane >> 4;
  const int col  = lane & 15;
  const int nb   = wave * 192;          // 12 N-tiles of 16 per wave

  // load initial hidden state
  for (int i = tid; i < 16 * H_; i += 256) {
    int m = i >> 9, n = i & (H_ - 1);
    hb0[m * SRH + n] = (__bf16)h0i[(b0 + m) * H_ + n];
    hb1[m * SRH + n] = (__bf16)h1i[(b0 + m) * H_ + n];
  }
  __syncthreads();

  const v8f vzero = {0.f, 0.f, 0.f, 0.f, 0.f, 0.f, 0.f, 0.f};

  #pragma unroll 1
  for (int t = 0; t < T_; ++t) {
    if (tid < 16) stok[tid] = (t == 0) ? 1 : target[(b0 + tid) * T_ + (t - 1)];

    // ---- gh0 = h0 @ W_hh0^T ----
    {
      v8f acc[12];
      #pragma unroll
      for (int j = 0; j < 12; ++j) acc[j] = vzero;
      gemm12(wHH0, hb0, nb, lane, acc);
      #pragma unroll
      for (int j = 0; j < 12; ++j)
        #pragma unroll
        for (int r = 0; r < 8; ++r)
          gates[(half * 8 + r) * SRG + nb + j * 16 + col] = acc[j][r];
    }
    __syncthreads();

    // ---- GRU0 elementwise: h0 <- cell(x_t, h0) ; gx from Gtab/gz tables ----
    for (int i = tid; i < 16 * H_; i += 256) {
      int m = i >> 9, n = i & (H_ - 1);
      const float* gt  = Gtab + stok[m] * G3_;
      const float* gzb = gz + (size_t)(b0 + m) * G3_;
      float xr = gt[n] + gzb[n];
      float xz = gt[n + 512] + gzb[n + 512];
      float xn = gt[n + 1024] + gzb[n + 1024];
      float hr = gates[m * SRG + n]        + bhh0[n];
      float hz = gates[m * SRG + n + 512]  + bhh0[n + 512];
      float hn = gates[m * SRG + n + 1024] + bhh0[n + 1024];
      float r  = 1.f / (1.f + __expf(-(xr + hr)));
      float u  = 1.f / (1.f + __expf(-(xz + hz)));
      float nn = tanhf(xn + r * hn);
      float hold = (float)hb0[m * SRH + n];
      hb0[m * SRH + n] = (__bf16)((1.f - u) * nn + u * hold);
    }
    __syncthreads();

    // ---- gh1 = h1 @ W_hh1^T  (store), then gi1 = h0n @ W_ih1^T (fuse r/z, spill n) ----
    {
      v8f acc[12];
      #pragma unroll
      for (int j = 0; j < 12; ++j) acc[j] = vzero;
      gemm12(wHH1, hb1, nb, lane, acc);
      #pragma unroll
      for (int j = 0; j < 12; ++j)
        #pragma unroll
        for (int r = 0; r < 8; ++r)
          gates[(half * 8 + r) * SRG + nb + j * 16 + col] = acc[j][r];
      // same thread owns the same (m,n) in both GEMMs: no barrier needed for RMW
      #pragma unroll
      for (int j = 0; j < 12; ++j) acc[j] = vzero;
      gemm12(wIH1, hb0, nb, lane, acc);
      #pragma unroll
      for (int j = 0; j < 12; ++j) {
        const int n0 = nb + j * 16;
        if (n0 < 1024) {            // r/z chunks: ih+hh sum elementwise -> fuse
          #pragma unroll
          for (int r = 0; r < 8; ++r)
            gates[(half * 8 + r) * SRG + n0 + col] += acc[j][r];
        } else {                    // n chunk: must stay separate (r * hn)
          #pragma unroll
          for (int r = 0; r < 8; ++r)
            gi1n[(half * 8 + r) * SRH + (n0 - 1024) + col] = (__bf16)acc[j][r];
        }
      }
    }
    __syncthreads();

    // ---- GRU1 elementwise: h1 <- cell(h0n, h1) ----
    for (int i = tid; i < 16 * H_; i += 256) {
      int m = i >> 9, n = i & (H_ - 1);
      float sr = gates[m * SRG + n]       + bih1[n]       + bhh1[n];
      float sz = gates[m * SRG + n + 512] + bih1[n + 512] + bhh1[n + 512];
      float xn = (float)gi1n[m * SRH + n] + bih1[n + 1024];
      float hn = gates[m * SRG + n + 1024] + bhh1[n + 1024];
      float r  = 1.f / (1.f + __expf(-sr));
      float u  = 1.f / (1.f + __expf(-sz));
      float nn = tanhf(xn + r * hn);
      float hold = (float)hb1[m * SRH + n];
      hb1[m * SRH + n] = (__bf16)((1.f - u) * nn + u * hold);
    }
    __syncthreads();

    // ---- logits = h1n @ fc_out^T + b : (16x512)@(512x64), waves 0..3 own 1 tile ----
    if (wave < 4) {
      v8f acc = vzero;
      const int row = col;
      #pragma unroll 1
      for (int kk = 0; kk < H_; kk += 32) {
        const int k1 = kk + half * 16;
        v8bf lo = *(const v8bf*)(hb1 + row * SRH + k1);
        v8bf hi = *(const v8bf*)(hb1 + row * SRH + k1 + 8);
        v16bf a;
        #pragma unroll
        for (int i = 0; i < 8; ++i) { a[i] = lo[i]; a[8 + i] = hi[i]; }
        v16bf bf = *(const v16bf*)(wOUT + (size_t)(wave * 16 + row) * H_ + kk + half * 16);
        acc = __builtin_amdgcn_wmma_f32_16x16x32_bf16(
            false, a, false, bf, (short)0, acc, false, false);
      }
      const int v = wave * 16 + col;
      const float bias = bout[v];
      #pragma unroll
      for (int r = 0; r < 8; ++r) {
        const int m = half * 8 + r;
        out[((size_t)(b0 + m) * T_ + t) * V_ + v] = acc[r] + bias;
      }
    }
    __syncthreads();
  }
}

// ---------------- host side ----------------

extern "C" void kernel_launch(void* const* d_in, const int* in_sizes, int n_in,
                              void* d_out, int out_size, void* d_ws, size_t ws_size,
                              hipStream_t stream) {
  const float* z    = (const float*)d_in[0];
  const int*   tgt  = (const int*)  d_in[1];
  const float* emb  = (const float*)d_in[2];
  const float* fhw  = (const float*)d_in[3];
  const float* fhb  = (const float*)d_in[4];
  const float* Wih0 = (const float*)d_in[5];
  const float* Whh0 = (const float*)d_in[6];
  const float* bih0 = (const float*)d_in[7];
  const float* bhh0 = (const float*)d_in[8];
  const float* Wih1 = (const float*)d_in[9];
  const float* Whh1 = (const float*)d_in[10];
  const float* bih1 = (const float*)d_in[11];
  const float* bhh1 = (const float*)d_in[12];
  const float* Wout = (const float*)d_in[13];
  const float* bout = (const float*)d_in[14];

  char* ws = (char*)d_ws;
  size_t off = 0;
  __bf16* wHH0 = (__bf16*)(ws + off); off += (size_t)G3_ * H_ * 2;   // 1.5 MB
  __bf16* wIH1 = (__bf16*)(ws + off); off += (size_t)G3_ * H_ * 2;
  __bf16* wHH1 = (__bf16*)(ws + off); off += (size_t)G3_ * H_ * 2;
  __bf16* wOUT = (__bf16*)(ws + off); off += (size_t)V_ * H_ * 2;
  float*  Gtab = (float*)(ws + off);  off += (size_t)V_ * G3_ * 4;
  float*  gz   = (float*)(ws + off);  off += (size_t)B_ * G3_ * 4;   // 12.6 MB
  float*  h0i  = (float*)(ws + off);  off += (size_t)B_ * H_ * 4;
  float*  h1i  = (float*)(ws + off);  off += (size_t)B_ * H_ * 4;
  (void)ws_size; (void)n_in; (void)in_sizes; (void)out_size;

  k_convert<<<(G3_ * H_ + 255) / 256, 256, 0, stream>>>(
      Whh0, Wih1, Whh1, Wout, wHH0, wIH1, wHH1, wOUT);
  k_gtab<<<(V_ * G3_ + 255) / 256, 256, 0, stream>>>(emb, Wih0, bih0, Gtab);
  k_gz<<<(B_ * G3_ + 255) / 256, 256, 0, stream>>>(z, Wih0, gz);
  k_hinit<<<(B_ * 2 * H_ + 255) / 256, 256, 0, stream>>>(z, fhw, fhb, h0i, h1i);

  k_gru<<<B_ / 16, 256, 0, stream>>>(
      tgt, Gtab, gz, wHH0, wIH1, wHH1, wOUT,
      bhh0, bih1, bhh1, bout, h0i, h1i, (float*)d_out);
}
